// pew_GRU_10170482557697
// MI455X (gfx1250) — matrix operand order repaired and checked by reading, not compiled, not run on hardware
//
#include <hip/hip_runtime.h>
#include <hip/hip_bf16.h>

// ---------------- problem constants ----------------
constexpr int Bn = 4096, Tn = 24, In = 128, Hn = 512, Wn = 32;
constexpr int BTn = Bn * Tn;                // 98304 rows

// ---------------- WMMA types / helpers ----------------
typedef __attribute__((ext_vector_type(16))) __bf16 v16bf;
typedef __attribute__((ext_vector_type(8)))  float  v8f;
typedef __attribute__((ext_vector_type(4)))  unsigned int u32x4;

union Frag16 { u32x4 u[2]; v16bf v; };

__device__ __forceinline__ v16bf load_frag16(const __hip_bfloat16* p) {
  Frag16 f;
  f.u[0] = *(const u32x4*)(p);
  f.u[1] = *(const u32x4*)(p + 8);
  return f.v;
}
// A-matrix 16x32 bf16 fragment: per-lane one contiguous 16-elem run at k = 16*half
__device__ __forceinline__ v16bf load_a_frag(const __hip_bfloat16* row, int kb, int half) {
  Frag16 f;
  f.u[0] = *(const u32x4*)(row + kb + half * 16);
  f.u[1] = *(const u32x4*)(row + kb + half * 16 + 8);
  return f.v;
}
__device__ __forceinline__ v16bf zero_frag() {
  Frag16 f;
  f.u[0] = u32x4{0u, 0u, 0u, 0u};
  f.u[1] = u32x4{0u, 0u, 0u, 0u};
  return f.v;
}
__device__ __forceinline__ v8f zero_acc() {
  return v8f{0.f, 0.f, 0.f, 0.f, 0.f, 0.f, 0.f, 0.f};
}
__device__ __forceinline__ v8f wmma_bf16(v16bf a, v16bf b, v8f c) {
  return __builtin_amdgcn_wmma_f32_16x16x32_bf16(false, a, false, b, (short)0, c, false, false);
}
__device__ __forceinline__ float sigmoidf_(float x) { return 1.0f / (1.0f + __expf(-x)); }

// ---------------- LDS-staged block-tile machinery ----------------
// Block tile: 128M x 64N, 4 waves stacked in M, all sharing one 64-row B tile
// staged in LDS in 64x128-K chunks (row stride 136 bf16 => optimal ds_load banking).
constexpr int LDS_STRIDE = 136;             // 128 + 8 pad (bf16 elements)

#define TILE_SETUP_LDS(NT_)                              \
  const int tid  = threadIdx.x;                          \
  const int lane = tid & 31;                             \
  const int wid  = tid >> 5;                             \
  const int bmt  = blockIdx.x / (NT_);                   \
  const int nt   = blockIdx.x % (NT_);                   \
  const int half = lane >> 4;                            \
  const int lm   = lane & 15;                            \
  const int mrow = bmt * 128 + wid * 32;                 \
  const int am0  = mrow + lm;                            \
  const int am1  = am0 + 16;

// cooperative copy of one 64-row x 128-K weight chunk into LDS (128 threads)
__device__ __forceinline__ void stage_b_chunk(const __hip_bfloat16* __restrict__ Wsrc, int ldk,
                                              unsigned short* LB, int tid) {
  const int ln   = tid >> 1;
  const int koff = (tid & 1) << 6;       // 0 or 64
  const __hip_bfloat16* s = Wsrc + (size_t)ln * ldk + koff;
  unsigned short* d = LB + ln * LDS_STRIDE + koff;
#pragma unroll
  for (int i = 0; i < 8; i++)
    *(u32x4*)(d + i * 8) = *(const u32x4*)(s + i * 8);   // ds_store_b128
}

// one 128-K chunk of WMMAs for the 32x64 wave tile, B fragments from LDS.
// v0/v1 predicate the A loads (shift boundary); WMMA runs at full EXEC.
__device__ __forceinline__ void wmma_chunk(const __hip_bfloat16* __restrict__ a0c,
                                           const __hip_bfloat16* __restrict__ a1c,
                                           const unsigned short* LB, int lm, int half,
                                           bool v0, bool v1,
                                           v8f acc0[4], v8f acc1[4]) {
  if (v0) __builtin_prefetch(a0c + 128, 0, 3);   // gfx1250 global_prefetch_b8 (speculative)
  if (v1) __builtin_prefetch(a1c + 128, 0, 3);
  const __hip_bfloat16* Bl = (const __hip_bfloat16*)LB + lm * LDS_STRIDE + half * 16;
#pragma unroll
  for (int ks = 0; ks < 4; ks++) {
    const int kb = ks * 32;
    v16bf a0 = zero_frag(), a1 = zero_frag();
    if (v0) a0 = load_a_frag(a0c, kb, half);
    if (v1) a1 = load_a_frag(a1c, kb, half);
#pragma unroll
    for (int j = 0; j < 4; j++) {
      v16bf b = load_frag16(Bl + j * 16 * LDS_STRIDE + kb);   // ds_load_b128
      acc0[j] = wmma_bf16(a0, b, acc0[j]);
      acc1[j] = wmma_bf16(a1, b, acc1[j]);
    }
  }
}

// per-wave 32x64 tile with arbitrary (mt,nt) mapping (used only by small-K g_e)
#define TILE_SETUP32(NT_)                                               \
  const int lane = threadIdx.x & 31;                                    \
  const int gw   = blockIdx.x * (blockDim.x >> 5) + (threadIdx.x >> 5); \
  const int mt   = gw / (NT_);                                          \
  const int nt   = gw % (NT_);                                          \
  const int half = lane >> 4;                                           \
  const int lm   = lane & 15;                                           \
  const int am0  = mt * 32 + lm;                                        \
  const int am1  = am0 + 16;

// ---------------- elementwise prep kernels ----------------
__global__ void k_f32_to_bf16(const float* __restrict__ s, __hip_bfloat16* __restrict__ d, int n) {
  int i = blockIdx.x * blockDim.x + threadIdx.x;
  if (i < n) d[i] = __float2bfloat16(s[i]);
}
__global__ void k_zero_bf16(__hip_bfloat16* __restrict__ d, int n) {
  int i = blockIdx.x * blockDim.x + threadIdx.x;
  if (i < n) d[i] = __float2bfloat16(0.0f);
}
// Wrz[1024][640] = [[w_rx|w_re],[w_zx|w_ze]]  (row n, col k; B stored as [N][K])
__global__ void k_build_wrz(const float* __restrict__ wrx, const float* __restrict__ wre,
                            const float* __restrict__ wzx, const float* __restrict__ wze,
                            __hip_bfloat16* __restrict__ d) {
  int i = blockIdx.x * blockDim.x + threadIdx.x;
  if (i >= 1024 * 640) return;
  int n = i / 640, k = i % 640;
  float v;
  if (n < Hn) v = (k < In) ? wrx[n * In + k] : wre[n * Hn + (k - In)];
  else        v = (k < In) ? wzx[(n - Hn) * In + k] : wze[(n - Hn) * Hn + (k - In)];
  d[i] = __float2bfloat16(v);
}
// Wo[512][384] = [w_d | w_w | w_m] along K
__global__ void k_build_wo(const float* __restrict__ wd, const float* __restrict__ ww,
                           const float* __restrict__ wm, __hip_bfloat16* __restrict__ d) {
  int i = blockIdx.x * blockDim.x + threadIdx.x;
  if (i >= Hn * 384) return;
  int n = i / 384, k = i % 384;
  int s = k >> 7, kk = k & 127;
  const float* src = (s == 0) ? wd : (s == 1) ? ww : wm;
  d[i] = __float2bfloat16(src[n * In + kk]);
}
// [a;b] stacked along N, each [rows][k]
__global__ void k_build_cat2(const float* __restrict__ a, const float* __restrict__ b,
                             __hip_bfloat16* __restrict__ d, int rows, int k) {
  int i = blockIdx.x * blockDim.x + threadIdx.x;
  if (i >= 2 * rows * k) return;
  int n = i / k, kk = i % k;
  float v = (n < rows) ? a[n * k + kk] : b[(n - rows) * k + kk];
  d[i] = __float2bfloat16(v);
}

// ---------------- phase 1: batched precompute GEMMs ----------------
// e = sigmoid(x_w @ w_e^T + b_e)   M=BT K=32 N=512, store bf16 (single K-step: no LDS)
__global__ void g_e_kernel(const __hip_bfloat16* __restrict__ xwb,
                           const __hip_bfloat16* __restrict__ We,
                           const float* __restrict__ be,
                           __hip_bfloat16* __restrict__ ebf) {
  TILE_SETUP32(Hn / 64)
  v8f acc0[4], acc1[4];
#pragma unroll
  for (int j = 0; j < 4; j++) { acc0[j] = zero_acc(); acc1[j] = zero_acc(); }
  const __hip_bfloat16* Wl = We + (size_t)(nt * 64 + lm) * Wn + half * 16;
  v16bf a0 = load_a_frag(xwb + (size_t)am0 * Wn, 0, half);
  v16bf a1 = load_a_frag(xwb + (size_t)am1 * Wn, 0, half);
#pragma unroll
  for (int j = 0; j < 4; j++) {
    v16bf b = load_frag16(Wl + (size_t)j * 16 * Wn);
    acc0[j] = wmma_bf16(a0, b, acc0[j]);
    acc1[j] = wmma_bf16(a1, b, acc1[j]);
  }
#pragma unroll
  for (int j = 0; j < 4; j++) {
    int n = nt * 64 + j * 16 + lm;
    float bias = be[n];
#pragma unroll
    for (int r = 0; r < 8; r++) {
      int mm0 = mt * 32 + r + 8 * half, mm1 = mm0 + 16;
      ebf[(size_t)mm0 * Hn + n] = __float2bfloat16(sigmoidf_(acc0[j][r] + bias));
      ebf[(size_t)mm1 * Hn + n] = __float2bfloat16(sigmoidf_(acc1[j][r] + bias));
    }
  }
}
// pre_rz = [x | e] @ Wrz^T + [b_r|b_z]   M=BT K=640 N=1024, fp32 out
__global__ void g_rz_kernel(const __hip_bfloat16* __restrict__ xb,
                            const __hip_bfloat16* __restrict__ ebf,
                            const __hip_bfloat16* __restrict__ Wrz,
                            const float* __restrict__ br, const float* __restrict__ bz,
                            float* __restrict__ prerz) {
  __shared__ unsigned short LB[64 * LDS_STRIDE];
  TILE_SETUP_LDS(1024 / 64)
  v8f acc0[4], acc1[4];
#pragma unroll
  for (int j = 0; j < 4; j++) { acc0[j] = zero_acc(); acc1[j] = zero_acc(); }
  for (int kc = 0; kc < 5; kc++) {
    if (kc) __syncthreads();
    stage_b_chunk(Wrz + (size_t)(nt * 64) * 640 + kc * 128, 640, LB, tid);
    __syncthreads();
    const __hip_bfloat16 *a0c, *a1c;
    if (kc == 0) { a0c = xb + (size_t)am0 * In;                  a1c = xb + (size_t)am1 * In; }
    else         { a0c = ebf + (size_t)am0 * Hn + (kc - 1) * 128; a1c = ebf + (size_t)am1 * Hn + (kc - 1) * 128; }
    wmma_chunk(a0c, a1c, LB, lm, half, true, true, acc0, acc1);
  }
#pragma unroll
  for (int j = 0; j < 4; j++) {
    int n = nt * 64 + j * 16 + lm;
    float bias = (n < Hn) ? br[n] : bz[n - Hn];
#pragma unroll
    for (int r = 0; r < 8; r++) {
      int mm0 = mrow + r + 8 * half, mm1 = mm0 + 16;
      prerz[(size_t)mm0 * 1024 + n] = acc0[j][r] + bias;
      prerz[(size_t)mm1 * 1024 + n] = acc1[j][r] + bias;
    }
  }
}
// pre_h = x @ w_hx^T + b_h   M=BT K=128 N=512
__global__ void g_h_kernel(const __hip_bfloat16* __restrict__ xb,
                           const __hip_bfloat16* __restrict__ Whx,
                           const float* __restrict__ bh,
                           float* __restrict__ preh) {
  __shared__ unsigned short LB[64 * LDS_STRIDE];
  TILE_SETUP_LDS(Hn / 64)
  v8f acc0[4], acc1[4];
#pragma unroll
  for (int j = 0; j < 4; j++) { acc0[j] = zero_acc(); acc1[j] = zero_acc(); }
  stage_b_chunk(Whx + (size_t)(nt * 64) * In, In, LB, tid);
  __syncthreads();
  wmma_chunk(xb + (size_t)am0 * In, xb + (size_t)am1 * In, LB, lm, half, true, true, acc0, acc1);
#pragma unroll
  for (int j = 0; j < 4; j++) {
    int n = nt * 64 + j * 16 + lm;
    float bias = bh[n];
#pragma unroll
    for (int r = 0; r < 8; r++) {
      int mm0 = mrow + r + 8 * half, mm1 = mm0 + 16;
      preh[(size_t)mm0 * Hn + n] = acc0[j][r] + bias;
      preh[(size_t)mm1 * Hn + n] = acc1[j][r] + bias;
    }
  }
}
// pre_o = x[-1]@w_d^T + x[-7]@w_w^T + x[-30]@w_m^T  (shifts along B), K=384
__global__ void g_o_kernel(const __hip_bfloat16* __restrict__ xb,
                           const __hip_bfloat16* __restrict__ Wo,
                           float* __restrict__ preo) {
  __shared__ unsigned short LB[64 * LDS_STRIDE];
  TILE_SETUP_LDS(Hn / 64)
  const int b0 = am0 / Tn, t0 = am0 - b0 * Tn;
  const int b1 = am1 / Tn, t1 = am1 - b1 * Tn;
  v8f acc0[4], acc1[4];
#pragma unroll
  for (int j = 0; j < 4; j++) { acc0[j] = zero_acc(); acc1[j] = zero_acc(); }
  for (int s = 0; s < 3; s++) {
    if (s) __syncthreads();
    stage_b_chunk(Wo + (size_t)(nt * 64) * 384 + s * 128, 384, LB, tid);
    __syncthreads();
    const int shift = (s == 0) ? 1 : (s == 1) ? 7 : 30;
    const int br0 = b0 - shift, br1 = b1 - shift;
    wmma_chunk(xb + (size_t)(br0 * Tn + t0) * In, xb + (size_t)(br1 * Tn + t1) * In,
               LB, lm, half, br0 >= 0, br1 >= 0, acc0, acc1);
  }
#pragma unroll
  for (int j = 0; j < 4; j++) {
    int n = nt * 64 + j * 16 + lm;
#pragma unroll
    for (int r = 0; r < 8; r++) {
      int mm0 = mrow + r + 8 * half, mm1 = mm0 + 16;
      preo[(size_t)mm0 * Hn + n] = acc0[j][r];
      preo[(size_t)mm1 * Hn + n] = acc1[j][r];
    }
  }
}

// ---------------- phase 2: recurrence (per time-step, 3 stream-ordered stages) ----------------
// stage 1: h_o = sigmoid(pre_o[:,t,:] + h @ w_t^T)   M=B K=512 N=512
__global__ void r1_kernel(const __hip_bfloat16* __restrict__ hb,
                          const __hip_bfloat16* __restrict__ Wt,
                          const float* __restrict__ preo,
                          float* __restrict__ ho, __hip_bfloat16* __restrict__ hob, int t) {
  __shared__ unsigned short LB[64 * LDS_STRIDE];
  TILE_SETUP_LDS(Hn / 64)
  v8f acc0[4], acc1[4];
#pragma unroll
  for (int j = 0; j < 4; j++) { acc0[j] = zero_acc(); acc1[j] = zero_acc(); }
  for (int kc = 0; kc < 4; kc++) {
    if (kc) __syncthreads();
    stage_b_chunk(Wt + (size_t)(nt * 64) * Hn + kc * 128, Hn, LB, tid);
    __syncthreads();
    wmma_chunk(hb + (size_t)am0 * Hn + kc * 128, hb + (size_t)am1 * Hn + kc * 128,
               LB, lm, half, true, true, acc0, acc1);
  }
#pragma unroll
  for (int j = 0; j < 4; j++) {
    int n = nt * 64 + j * 16 + lm;
#pragma unroll
    for (int r = 0; r < 8; r++) {
      int mm0 = mrow + r + 8 * half, mm1 = mm0 + 16;   // batch indices
      float v0 = sigmoidf_(acc0[j][r] + preo[((size_t)mm0 * Tn + t) * Hn + n]);
      float v1 = sigmoidf_(acc1[j][r] + preo[((size_t)mm1 * Tn + t) * Hn + n]);
      ho[(size_t)mm0 * Hn + n] = v0;  hob[(size_t)mm0 * Hn + n] = __float2bfloat16(v0);
      ho[(size_t)mm1 * Hn + n] = v1;  hob[(size_t)mm1 * Hn + n] = __float2bfloat16(v1);
    }
  }
}
// stage 2: [r|z] = sigmoid(pre_rz[:,t,:] + h_o @ [w_rh|w_zh]^T); emit (r*h_o) bf16 and z f32
__global__ void r2_kernel(const __hip_bfloat16* __restrict__ hob,
                          const __hip_bfloat16* __restrict__ Wrzh,
                          const float* __restrict__ prerz,
                          const float* __restrict__ ho,
                          __hip_bfloat16* __restrict__ rhob, float* __restrict__ zf, int t) {
  __shared__ unsigned short LB[64 * LDS_STRIDE];
  TILE_SETUP_LDS(1024 / 64)
  v8f acc0[4], acc1[4];
#pragma unroll
  for (int j = 0; j < 4; j++) { acc0[j] = zero_acc(); acc1[j] = zero_acc(); }
  for (int kc = 0; kc < 4; kc++) {
    if (kc) __syncthreads();
    stage_b_chunk(Wrzh + (size_t)(nt * 64) * Hn + kc * 128, Hn, LB, tid);
    __syncthreads();
    wmma_chunk(hob + (size_t)am0 * Hn + kc * 128, hob + (size_t)am1 * Hn + kc * 128,
               LB, lm, half, true, true, acc0, acc1);
  }
#pragma unroll
  for (int j = 0; j < 4; j++) {
    int n = nt * 64 + j * 16 + lm;
#pragma unroll
    for (int r = 0; r < 8; r++) {
      int mm0 = mrow + r + 8 * half, mm1 = mm0 + 16;
      float v0 = sigmoidf_(acc0[j][r] + prerz[((size_t)mm0 * Tn + t) * 1024 + n]);
      float v1 = sigmoidf_(acc1[j][r] + prerz[((size_t)mm1 * Tn + t) * 1024 + n]);
      if (n < Hn) {
        rhob[(size_t)mm0 * Hn + n] = __float2bfloat16(v0 * ho[(size_t)mm0 * Hn + n]);
        rhob[(size_t)mm1 * Hn + n] = __float2bfloat16(v1 * ho[(size_t)mm1 * Hn + n]);
      } else {
        zf[(size_t)mm0 * Hn + (n - Hn)] = v0;
        zf[(size_t)mm1 * Hn + (n - Hn)] = v1;
      }
    }
  }
}
// stage 3: h_tilde = tanh(pre_h + (r*h_o) @ w_hh^T); h = (1-z)h_o + z h_tilde -> out & next-step bf16
__global__ void r3_kernel(const __hip_bfloat16* __restrict__ rhob,
                          const __hip_bfloat16* __restrict__ Whh,
                          const float* __restrict__ preh,
                          const float* __restrict__ ho, const float* __restrict__ zf,
                          float* __restrict__ out, __hip_bfloat16* __restrict__ hb, int t) {
  __shared__ unsigned short LB[64 * LDS_STRIDE];
  TILE_SETUP_LDS(Hn / 64)
  v8f acc0[4], acc1[4];
#pragma unroll
  for (int j = 0; j < 4; j++) { acc0[j] = zero_acc(); acc1[j] = zero_acc(); }
  for (int kc = 0; kc < 4; kc++) {
    if (kc) __syncthreads();
    stage_b_chunk(Whh + (size_t)(nt * 64) * Hn + kc * 128, Hn, LB, tid);
    __syncthreads();
    wmma_chunk(rhob + (size_t)am0 * Hn + kc * 128, rhob + (size_t)am1 * Hn + kc * 128,
               LB, lm, half, true, true, acc0, acc1);
  }
#pragma unroll
  for (int j = 0; j < 4; j++) {
    int n = nt * 64 + j * 16 + lm;
#pragma unroll
    for (int r = 0; r < 8; r++) {
      int mm0 = mrow + r + 8 * half, mm1 = mm0 + 16;
#pragma unroll
      for (int g = 0; g < 2; g++) {
        int mm = g ? mm1 : mm0;
        float a = g ? acc1[j][r] : acc0[j][r];
        float ht = tanhf(a + preh[((size_t)mm * Tn + t) * Hn + n]);
        float z  = zf[(size_t)mm * Hn + n];
        float h0 = ho[(size_t)mm * Hn + n];
        float hn = (1.0f - z) * h0 + z * ht;
        out[((size_t)mm * Tn + t) * Hn + n] = hn;
        hb[(size_t)mm * Hn + n] = __float2bfloat16(hn);
      }
    }
  }
}

// ---------------- host ----------------
extern "C" void kernel_launch(void* const* d_in, const int* in_sizes, int n_in,
                              void* d_out, int out_size, void* d_ws, size_t ws_size,
                              hipStream_t stream) {
  const float* x_input   = (const float*)d_in[0];
  const float* x_weather = (const float*)d_in[1];
  const float* w_rx = (const float*)d_in[2];
  const float* w_rh = (const float*)d_in[3];
  const float* w_re = (const float*)d_in[4];
  const float* b_r  = (const float*)d_in[5];
  const float* w_zx = (const float*)d_in[6];
  const float* w_zh = (const float*)d_in[7];
  const float* w_ze = (const float*)d_in[8];
  const float* b_z  = (const float*)d_in[9];
  const float* w_hx = (const float*)d_in[10];
  const float* w_hh = (const float*)d_in[11];
  const float* b_h  = (const float*)d_in[12];
  const float* w_d  = (const float*)d_in[13];
  const float* w_w  = (const float*)d_in[14];
  const float* w_m  = (const float*)d_in[15];
  const float* w_t  = (const float*)d_in[16];
  const float* w_e  = (const float*)d_in[17];
  const float* b_e  = (const float*)d_in[18];
  float* out = (float*)d_out;

  char* base = (char*)d_ws;
  size_t off = 0;
  auto alloc = [&](size_t bytes) -> void* {
    void* p = base + off;
    off = (off + bytes + 255) & ~(size_t)255;
    return p;
  };
  __hip_bfloat16* xb    = (__hip_bfloat16*)alloc((size_t)BTn * In * 2);
  __hip_bfloat16* xwb   = (__hip_bfloat16*)alloc((size_t)BTn * Wn * 2);
  __hip_bfloat16* ebf   = (__hip_bfloat16*)alloc((size_t)BTn * Hn * 2);
  __hip_bfloat16* Wrz   = (__hip_bfloat16*)alloc((size_t)1024 * 640 * 2);
  __hip_bfloat16* Whxb  = (__hip_bfloat16*)alloc((size_t)Hn * In * 2);
  __hip_bfloat16* Wob   = (__hip_bfloat16*)alloc((size_t)Hn * 384 * 2);
  __hip_bfloat16* Wtb   = (__hip_bfloat16*)alloc((size_t)Hn * Hn * 2);
  __hip_bfloat16* Web   = (__hip_bfloat16*)alloc((size_t)Hn * Wn * 2);
  __hip_bfloat16* Wrzhb = (__hip_bfloat16*)alloc((size_t)1024 * Hn * 2);
  __hip_bfloat16* Whhb  = (__hip_bfloat16*)alloc((size_t)Hn * Hn * 2);
  float* prerz = (float*)alloc((size_t)BTn * 1024 * 4);
  float* preh  = (float*)alloc((size_t)BTn * Hn * 4);
  float* preo  = (float*)alloc((size_t)BTn * Hn * 4);
  float* ho    = (float*)alloc((size_t)Bn * Hn * 4);
  float* zf    = (float*)alloc((size_t)Bn * Hn * 4);
  __hip_bfloat16* hob  = (__hip_bfloat16*)alloc((size_t)Bn * Hn * 2);
  __hip_bfloat16* rhob = (__hip_bfloat16*)alloc((size_t)Bn * Hn * 2);
  __hip_bfloat16* hb   = (__hip_bfloat16*)alloc((size_t)Bn * Hn * 2);
  (void)ws_size; (void)in_sizes; (void)n_in; (void)out_size;

  dim3 cb(256);
  auto cg = [](long n) { return dim3((unsigned)((n + 255) / 256)); };

  // --- prep: conversions & weight concats (bf16, [N][K] layout) ---
  k_f32_to_bf16<<<cg((long)BTn * In), cb, 0, stream>>>(x_input, xb, BTn * In);
  k_f32_to_bf16<<<cg((long)BTn * Wn), cb, 0, stream>>>(x_weather, xwb, BTn * Wn);
  k_f32_to_bf16<<<cg(Hn * In), cb, 0, stream>>>(w_hx, Whxb, Hn * In);
  k_f32_to_bf16<<<cg(Hn * Hn), cb, 0, stream>>>(w_t, Wtb, Hn * Hn);
  k_f32_to_bf16<<<cg(Hn * Hn), cb, 0, stream>>>(w_hh, Whhb, Hn * Hn);
  k_f32_to_bf16<<<cg(Hn * Wn), cb, 0, stream>>>(w_e, Web, Hn * Wn);
  k_build_wrz<<<cg(1024 * 640), cb, 0, stream>>>(w_rx, w_re, w_zx, w_ze, Wrz);
  k_build_wo<<<cg(Hn * 384), cb, 0, stream>>>(w_d, w_w, w_m, Wob);
  k_build_cat2<<<cg(1024 * Hn), cb, 0, stream>>>(w_rh, w_zh, Wrzhb, Hn, Hn);
  k_zero_bf16<<<cg(Bn * Hn), cb, 0, stream>>>(hb, Bn * Hn);

  // --- phase 1: batched WMMA GEMMs ---
  dim3 gb(128);
  // LDS-staged kernels: block tile 128M x 64N
  auto lds_grid = [](long M, long N) { return dim3((unsigned)((M / 128) * (N / 64))); };
  // g_e: per-wave 32x64 tiles, 4 waves/block
  auto g32_grid = [](long M, long N) { return dim3((unsigned)((M / 32) * (N / 64) / 4)); };
  g_e_kernel <<<g32_grid(BTn, Hn),   gb, 0, stream>>>(xwb, Web, b_e, ebf);
  g_rz_kernel<<<lds_grid(BTn, 1024), gb, 0, stream>>>(xb, ebf, Wrz, b_r, b_z, prerz);
  g_h_kernel <<<lds_grid(BTn, Hn),   gb, 0, stream>>>(xb, Whxb, b_h, preh);
  g_o_kernel <<<lds_grid(BTn, Hn),   gb, 0, stream>>>(xb, Wob, preo);

  // --- phase 2: recurrence, 24 steps x 3 dependent stages (stream order = dependency) ---
  for (int t = 0; t < Tn; t++) {
    r1_kernel<<<lds_grid(Bn, Hn),   gb, 0, stream>>>(hb, Wtb, preo, ho, hob, t);
    r2_kernel<<<lds_grid(Bn, 1024), gb, 0, stream>>>(hob, Wrzhb, prerz, ho, rhob, zf, t);
    r3_kernel<<<lds_grid(Bn, Hn),   gb, 0, stream>>>(rhob, Whhb, preh, ho, zf, out, hb, t);
  }
}
